// DeformableFusion_45243185496283
// MI455X (gfx1250) — compile-verified, matrix operand order checked by reading, not run on hardware
//
#include <hip/hip_runtime.h>
#include <math.h>

// ---------------------------------------------------------------------------
// DeformableFusion for MI455X (gfx1250): all convs as f16 WMMA implicit GEMM.
// conv_gemm<NIN,KSIZE>: K=64 stages, double-buffered LDS, async global->LDS.
// dcn_gemm           : fused bilinear im2col + WMMA, double-buffered K=32.
// ---------------------------------------------------------------------------

typedef __attribute__((ext_vector_type(16))) _Float16 v16h;
typedef __attribute__((ext_vector_type(8)))  float    v8f;
typedef __attribute__((ext_vector_type(8)))  _Float16 h8;

#define HW_   16384          // 128*128
#define MTOT_ 32768          // B(2) * 128 * 128

// ---- WMMA micro-tile, LDS row stride 64 (K=64 stage, two 32-deep chunks).
// A-fragment layout (ISA 7.12.2, 16-bit A 16x32): lane<16 -> K {0..7,16..23},
// lane>=16 -> K {8..15,24..31}; row = M (A) / N (B, stored [n][k]).
__device__ __forceinline__ void wmma_k64(const _Float16* __restrict__ As,
                                         const _Float16* __restrict__ Bs,
                                         int wm, int wn, int lane,
                                         v8f& acc0, v8f& acc1) {
  const int l = lane & 15;
#pragma unroll
  for (int kb = 0; kb < 64; kb += 32) {
    const int k0 = kb + (lane >> 4) * 8;
    union U { v16h v; h8 h[2]; };
    U a, b0, b1;
    const _Float16* ar  = As + (wm * 16 + l) * 64;
    a.h[0]  = *(const h8*)(ar + k0);
    a.h[1]  = *(const h8*)(ar + k0 + 16);
    const _Float16* br0 = Bs + (wn * 32 + l) * 64;
    b0.h[0] = *(const h8*)(br0 + k0);
    b0.h[1] = *(const h8*)(br0 + k0 + 16);
    const _Float16* br1 = Bs + (wn * 32 + 16 + l) * 64;
    b1.h[0] = *(const h8*)(br1 + k0);
    b1.h[1] = *(const h8*)(br1 + k0 + 16);
    acc0 = __builtin_amdgcn_wmma_f32_16x16x32_f16(false, a.v, false, b0.v,
                                                  (short)0, acc0, false, false);
    acc1 = __builtin_amdgcn_wmma_f32_16x16x32_f16(false, a.v, false, b1.v,
                                                  (short)0, acc1, false, false);
  }
}

// ---- same, LDS row stride 32 (DCN kernel, K=32 stage)
__device__ __forceinline__ void wmma_k32(const _Float16* __restrict__ As,
                                         const _Float16* __restrict__ Bs,
                                         int wm, int wn, int lane,
                                         v8f& acc0, v8f& acc1) {
  const int l  = lane & 15;
  const int k0 = (lane >> 4) * 8;
  union U { v16h v; h8 h[2]; };
  U a, b0, b1;
  const _Float16* ar  = As + (wm * 16 + l) * 32;
  a.h[0]  = *(const h8*)(ar + k0);
  a.h[1]  = *(const h8*)(ar + k0 + 16);
  const _Float16* br0 = Bs + (wn * 32 + l) * 32;
  b0.h[0] = *(const h8*)(br0 + k0);
  b0.h[1] = *(const h8*)(br0 + k0 + 16);
  const _Float16* br1 = Bs + (wn * 32 + 16 + l) * 32;
  b1.h[0] = *(const h8*)(br1 + k0);
  b1.h[1] = *(const h8*)(br1 + k0 + 16);
  acc0 = __builtin_amdgcn_wmma_f32_16x16x32_f16(false, a.v, false, b0.v,
                                                (short)0, acc0, false, false);
  acc1 = __builtin_amdgcn_wmma_f32_16x16x32_f16(false, a.v, false, b1.v,
                                                (short)0, acc1, false, false);
}

// ---- async global -> LDS, 16B each, LDS/mem share the instruction offset
__device__ __forceinline__ void async_b128(unsigned lds_addr,
                                           unsigned long gaddr) {
  asm volatile("global_load_async_to_lds_b128 %0, %1, off"
               :: "v"(lds_addr), "v"(gaddr) : "memory");
}
__device__ __forceinline__ void async_b128_o16(unsigned lds_addr,
                                               unsigned long gaddr) {
  asm volatile("global_load_async_to_lds_b128 %0, %1, off offset:16"
               :: "v"(lds_addr), "v"(gaddr) : "memory");
}
__device__ __forceinline__ void wait_async0() {
  asm volatile("s_wait_asynccnt 0" ::: "memory");
}

// ---- weight pack: w[Cout][Cin][KK] f32 -> wp[kp][CoutPad][Cin] f16 (pad=0)
__global__ __launch_bounds__(256)
void pack_w(const float* __restrict__ w, _Float16* __restrict__ wp,
            int Cout, int CoutPad, int Cin, int KKk, long total) {
  long idx = (long)blockIdx.x * 256 + threadIdx.x;
  if (idx >= total) return;
  long per_kp = (long)CoutPad * Cin;
  int  kp = (int)(idx / per_kp);
  long r  = idx - (long)kp * per_kp;
  int  co = (int)(r / Cin);
  int  ci = (int)(r - (long)co * Cin);
  float v = (co < Cout) ? w[((long)co * Cin + ci) * KKk + kp] : 0.0f;
  wp[idx] = (_Float16)v;
}

// ---- NCHW f32 -> NHWC f16 (C=256, H=W=128)
__global__ __launch_bounds__(256)
void to_nhwc_f16(const float* __restrict__ in, _Float16* __restrict__ out) {
  long idx = (long)blockIdx.x * 256 + threadIdx.x;   // over MTOT_*256
  int  m = (int)(idx >> 8);
  int  c = (int)(idx & 255);
  int  b = m >> 14;
  int  rr = m & 16383;
  out[idx] = (_Float16)in[(((long)b * 256 + c) << 14) + rr];
}

// ---- generic conv as implicit GEMM (64 pixels x 64 out-ch per block).
// NIN NHWC-f16 inputs of 256 ch each (concat along C), KSIZE in {1,3}.
// Compile-time NIN/KSIZE keep the stage address math division-free.
// mode: 0 = f16 NHWC (stride 256), 1 = f32 NHWC (stride Nout), 2 = f32 NCHW.
template <int NIN, int KSIZE>
__global__ __launch_bounds__(256)
void conv_gemm(const _Float16* __restrict__ in0,
               const _Float16* __restrict__ in1,
               const _Float16* __restrict__ in2,
               const _Float16* __restrict__ wp,
               const float* __restrict__ bias,
               _Float16* __restrict__ out16,
               float* __restrict__ out32,
               int Nout, int NoutPad, int relu, int mode) {
  constexpr int Cin = NIN * 256;
  constexpr int KKk = KSIZE * KSIZE;
  constexpr int pad = KSIZE / 2;
  constexpr int nck = Cin / 64;      // K stages per kernel tap
  constexpr int S   = KKk * nck;     // total K stages

  __shared__ __align__(16) _Float16 As[2][64 * 64];   // 2 x 8KB
  __shared__ __align__(16) _Float16 Bs[2][64 * 64];   // 2 x 8KB

  const int tid  = threadIdx.x;
  const int lane = tid & 31;
  const int wave = tid >> 5;
  const int wm = wave & 3;           // 4 waves along M
  const int wn = wave >> 2;          // 2 waves along N
  const long mtile = (long)blockIdx.x * 64;
  const int  ntile = blockIdx.y * 64;

  v8f acc0 = {}, acc1 = {};

  // staging coords: thread t moves 16 contiguous f16 of one row (2x b128)
  const int rp = tid >> 2;           // 0..63 (row in A and in B)
  const int kc = (tid & 3) * 16;     // 0,16,32,48 within 64-wide K stage
  const long m = mtile + rp;
  const int  b = (int)(m >> 14);
  const int  rr = (int)(m & 16383);
  const int  y = rr >> 7, x = rr & 127;

  auto stage = [&](int s, int bi) {
    const int kp = s / nck;                     // constexpr divisor
    const int cb = (s - kp * nck) << 6;         // channel base of this stage
    const int sy = y + kp / KSIZE - pad;
    const int sx = x + kp % KSIZE - pad;
    const bool valid = (sy >= 0) && (sy < 128) && (sx >= 0) && (sx < 128);
    const int cg  = cb + kc;                    // global channel of my chunk
    const int sel = cg >> 8;
    const _Float16* src = (sel == 0) ? in0 : ((sel == 1) ? in1 : in2);
    const unsigned lds_a = (unsigned)(unsigned long)&As[bi][rp * 64 + kc];
    if (valid) {
      const long spix = ((long)b << 14) + (long)sy * 128 + sx;
      const unsigned long ga =
          (unsigned long)(src + spix * 256 + (cg & 255));
      async_b128(lds_a, ga);
      async_b128_o16(lds_a, ga);
    } else {
      h8 z = {};
      *(h8*)&As[bi][rp * 64 + kc]     = z;
      *(h8*)&As[bi][rp * 64 + kc + 8] = z;
    }
    const unsigned lds_b = (unsigned)(unsigned long)&Bs[bi][rp * 64 + kc];
    const unsigned long gb = (unsigned long)(
        wp + ((long)kp * NoutPad + ntile + rp) * Cin + cg);
    async_b128(lds_b, gb);
    async_b128_o16(lds_b, gb);
  };

  // prologue
  stage(0, 0);
  wait_async0();
  __syncthreads();

  int buf = 0;
  for (int s = 0; s < S; ++s) {
    if (s + 1 < S) stage(s + 1, buf ^ 1);       // DMA next stage
    wmma_k64(As[buf], Bs[buf], wm, wn, lane, acc0, acc1);  // 4 WMMA/wave
    wait_async0();
    __syncthreads();
    buf ^= 1;
  }

  // C/D layout (ISA 7.12.2): lane<16 -> M=vgpr_i, lane>=16 -> M=8+i; N=lane&15
  const int l  = lane & 15;
  const int mb = (lane < 16) ? 0 : 8;
  const int n0 = ntile + wn * 32 + l;
  const int n1 = n0 + 16;
  const float bi0 = (bias && n0 < Nout) ? bias[n0] : 0.0f;
  const float bi1 = (bias && n1 < Nout) ? bias[n1] : 0.0f;
#pragma unroll
  for (int i = 0; i < 8; ++i) {
    const long mg = mtile + wm * 16 + mb + i;
    float v0 = acc0[i] + bi0;
    float v1 = acc1[i] + bi1;
    if (relu) { v0 = fmaxf(v0, 0.0f); v1 = fmaxf(v1, 0.0f); }
    const int bb = (int)(mg >> 14);
    const int rm = (int)(mg & 16383);
    if (n0 < Nout) {
      if (mode == 0)      out16[mg * 256 + n0] = (_Float16)v0;
      else if (mode == 1) out32[mg * Nout + n0] = v0;
      else                out32[(((long)bb * 256 + n0) << 14) + rm] = v0;
    }
    if (n1 < Nout) {
      if (mode == 0)      out16[mg * 256 + n1] = (_Float16)v1;
      else if (mode == 1) out32[mg * Nout + n1] = v1;
      else                out32[(((long)bb * 256 + n1) << 14) + rm] = v1;
    }
  }
}

// ---- deformable conv: fused bilinear im2col (A-tile) + WMMA GEMM core.
// Double-buffered K=32 stages, pipelined within each kernel tap so the
// bilinear coordinates stay hoisted per tap.
// ftn: NHWC f16 (256 ch). off: NHWC f32, stride 36, channels choff+kp*2{dy,dx}.
__global__ __launch_bounds__(256)
void dcn_gemm(const _Float16* __restrict__ ftn,
              const float* __restrict__ off, int choff,
              const _Float16* __restrict__ wp,
              _Float16* __restrict__ out16) {
  __shared__ __align__(16) _Float16 As[2][64 * 32];   // 2 x 4KB
  __shared__ __align__(16) _Float16 Bs[2][64 * 32];   // 2 x 4KB

  const int tid  = threadIdx.x;
  const int lane = tid & 31;
  const int wave = tid >> 5;
  const int wm = wave & 3, wn = wave >> 2;
  const long mtile = (long)blockIdx.x * 64;
  const int  ntile = blockIdx.y * 64;

  v8f acc0 = {}, acc1 = {};

  const int p  = tid >> 2;
  const int cc = (tid & 3) * 8;
  const long m = mtile + p;
  const int  b = (int)(m >> 14);
  const int  rr = (int)(m & 16383);
  const int  y = rr >> 7, x = rr & 127;
  const long base = (long)b << 14;

  for (int kp = 0; kp < 9; ++kp) {
    // per-thread sample coordinates for pixel p at kernel tap kp (hoisted)
    const float dy = off[m * 36 + choff + kp * 2];
    const float dx = off[m * 36 + choff + kp * 2 + 1];
    const float ys = (float)y + (float)(kp / 3 - 1) + dy;
    const float xs = (float)x + (float)(kp % 3 - 1) + dx;
    const float y0f = floorf(ys), x0f = floorf(xs);
    const float fy = ys - y0f, fx = xs - x0f;
    const int iy0 = (int)y0f, ix0 = (int)x0f;
    const int iy1 = iy0 + 1,  ix1 = ix0 + 1;
    const float w00 = (1.0f - fy) * (1.0f - fx);
    const float w01 = (1.0f - fy) * fx;
    const float w10 = fy * (1.0f - fx);
    const float w11 = fy * fx;
    const bool v00 = (iy0 >= 0) && (iy0 < 128) && (ix0 >= 0) && (ix0 < 128);
    const bool v01 = (iy0 >= 0) && (iy0 < 128) && (ix1 >= 0) && (ix1 < 128);
    const bool v10 = (iy1 >= 0) && (iy1 < 128) && (ix0 >= 0) && (ix0 < 128);
    const bool v11 = (iy1 >= 0) && (iy1 < 128) && (ix1 >= 0) && (ix1 < 128);

    auto stage = [&](int c0, int bi) {
      float vals[8] = {0, 0, 0, 0, 0, 0, 0, 0};
      const long cb = (long)(c0 + cc);
      if (v00) {
        h8 t = *(const h8*)(ftn + (base + (long)iy0 * 128 + ix0) * 256 + cb);
#pragma unroll
        for (int j = 0; j < 8; ++j) vals[j] += w00 * (float)t[j];
      }
      if (v01) {
        h8 t = *(const h8*)(ftn + (base + (long)iy0 * 128 + ix1) * 256 + cb);
#pragma unroll
        for (int j = 0; j < 8; ++j) vals[j] += w01 * (float)t[j];
      }
      if (v10) {
        h8 t = *(const h8*)(ftn + (base + (long)iy1 * 128 + ix0) * 256 + cb);
#pragma unroll
        for (int j = 0; j < 8; ++j) vals[j] += w10 * (float)t[j];
      }
      if (v11) {
        h8 t = *(const h8*)(ftn + (base + (long)iy1 * 128 + ix1) * 256 + cb);
#pragma unroll
        for (int j = 0; j < 8; ++j) vals[j] += w11 * (float)t[j];
      }
      h8 rv;
#pragma unroll
      for (int j = 0; j < 8; ++j) rv[j] = (_Float16)vals[j];
      *(h8*)&As[bi][p * 32 + cc] = rv;
      const long woff = ((long)kp * 256 + ntile + p) * 256 + c0 + cc;
      *(h8*)&Bs[bi][p * 32 + cc] = *(const h8*)(wp + woff);
      __builtin_prefetch(wp + woff + 32, 0, 1);   // global_prefetch_b8
    };

    // pipelined K stages within this tap (8 stages of K=32)
    stage(0, 0);
    __syncthreads();
    int buf = 0;
    for (int i = 0; i < 8; ++i) {
      if (i + 1 < 8) stage((i + 1) * 32, buf ^ 1);
      wmma_k32(As[buf], Bs[buf], wm, wn, lane, acc0, acc1);
      __syncthreads();
      buf ^= 1;
    }
  }

  const int l  = lane & 15;
  const int mb = (lane < 16) ? 0 : 8;
  const int n0 = ntile + wn * 32 + l;
  const int n1 = n0 + 16;
#pragma unroll
  for (int i = 0; i < 8; ++i) {
    const long mg = mtile + wm * 16 + mb + i;
    out16[mg * 256 + n0] = (_Float16)acc0[i];
    out16[mg * 256 + n1] = (_Float16)acc1[i];
  }
}

// ---------------------------------------------------------------------------
extern "C" void kernel_launch(void* const* d_in, const int* in_sizes, int n_in,
                              void* d_out, int out_size, void* d_ws,
                              size_t ws_size, hipStream_t stream) {
  (void)in_sizes; (void)n_in; (void)out_size; (void)ws_size;
  const float* ft1 = (const float*)d_in[0];
  const float* ft2 = (const float*)d_in[1];
  const float* ft3 = (const float*)d_in[2];
  const float* w1  = (const float*)d_in[3];
  const float* b1  = (const float*)d_in[4];
  const float* w2  = (const float*)d_in[5];
  const float* b2  = (const float*)d_in[6];
  const float* w3  = (const float*)d_in[7];
  const float* b3  = (const float*)d_in[8];
  const float* wd2 = (const float*)d_in[9];
  const float* wd3 = (const float*)d_in[10];
  const float* wfz = (const float*)d_in[11];
  const float* bfz = (const float*)d_in[12];
  float* out = (float*)d_out;

  char* ws = (char*)d_ws;
  size_t o = 0;
  auto take = [&](size_t bytes) -> char* {
    char* pp = ws + o;
    o += (bytes + 255) & ~(size_t)255;
    return pp;
  };
  const size_t actB = (size_t)MTOT_ * 256 * sizeof(_Float16);
  _Float16* ft1n = (_Float16*)take(actB);
  _Float16* ft2n = (_Float16*)take(actB);
  _Float16* ft3n = (_Float16*)take(actB);
  _Float16* h1b  = (_Float16*)take(actB);
  _Float16* h2b  = (_Float16*)take(actB);
  _Float16* al2  = (_Float16*)take(actB);
  _Float16* al3  = (_Float16*)take(actB);
  float*    offb = (float*)take((size_t)MTOT_ * 36 * sizeof(float));
  _Float16* wp1  = (_Float16*)take((size_t)9 * 256 * 768 * 2);
  _Float16* wp2  = (_Float16*)take((size_t)9 * 256 * 256 * 2);
  _Float16* wp3  = (_Float16*)take((size_t)9 * 64 * 256 * 2);
  _Float16* wpd2 = (_Float16*)take((size_t)9 * 256 * 256 * 2);
  _Float16* wpd3 = (_Float16*)take((size_t)9 * 256 * 256 * 2);
  _Float16* wpf  = (_Float16*)take((size_t)256 * 768 * 2);

  auto packL = [&](const float* w, _Float16* wpk, int Cout, int CoutPad,
                   int Cin, int KKk) {
    long total = (long)KKk * CoutPad * Cin;
    int blocks = (int)((total + 255) / 256);
    pack_w<<<blocks, 256, 0, stream>>>(w, wpk, Cout, CoutPad, Cin, KKk, total);
  };
  packL(w1,  wp1,  256, 256, 768, 9);
  packL(w2,  wp2,  256, 256, 256, 9);
  packL(w3,  wp3,   36,  64, 256, 9);
  packL(wd2, wpd2, 256, 256, 256, 9);
  packL(wd3, wpd3, 256, 256, 256, 9);
  packL(wfz, wpf,  256, 256, 768, 1);

  to_nhwc_f16<<<MTOT_, 256, 0, stream>>>(ft1, ft1n);
  to_nhwc_f16<<<MTOT_, 256, 0, stream>>>(ft2, ft2n);
  to_nhwc_f16<<<MTOT_, 256, 0, stream>>>(ft3, ft3n);

  dim3 g4(MTOT_ / 64, 4);
  dim3 g1(MTOT_ / 64, 1);
  // conv1: concat(ft1,ft2,ft3) 3x3 768->256 + ReLU
  conv_gemm<3, 3><<<g4, 256, 0, stream>>>(ft1n, ft2n, ft3n, wp1, b1, h1b,
                                          nullptr, 256, 256, 1, 0);
  // conv2: 3x3 256->256 + ReLU
  conv_gemm<1, 3><<<g4, 256, 0, stream>>>(h1b, nullptr, nullptr, wp2, b2, h2b,
                                          nullptr, 256, 256, 1, 0);
  // conv3: 3x3 256->36 offsets (f32 NHWC)
  conv_gemm<1, 3><<<g1, 256, 0, stream>>>(h2b, nullptr, nullptr, wp3, b3,
                                          nullptr, offb, 36, 64, 0, 1);
  // deformable convs
  dcn_gemm<<<g4, 256, 0, stream>>>(ft2n, offb, 0,  wpd2, al2);
  dcn_gemm<<<g4, 256, 0, stream>>>(ft3n, offb, 18, wpd3, al3);
  // fuse: 1x1 concat(ft1, al2, al3) 768->256 -> f32 NCHW output
  conv_gemm<3, 1><<<g4, 256, 0, stream>>>(ft1n, al2, al3, wpf, bfz, nullptr,
                                          out, 256, 256, 0, 2);
}
